// MultiHeadAttention_44444321579094
// MI455X (gfx1250) — matrix operand order repaired and checked by reading, not compiled, not run on hardware
//
#include <hip/hip_runtime.h>

typedef __bf16 bf16_t;
typedef __attribute__((ext_vector_type(16))) __bf16 v16bf;
typedef __attribute__((ext_vector_type(8)))  __bf16 v8bf;
typedef __attribute__((ext_vector_type(8)))  float  v8f;

#define S_LEN 4096
#define D_DIM 1024
#define H_CNT 16
#define HD_DIM 64

// ---------------------------------------------------------------------------
// WMMA helpers (CDNA5 16x16x32 bf16, f32 accumulate)
// ---------------------------------------------------------------------------
static __device__ __forceinline__ v8f wmma_bf16(v16bf a, v16bf b, v8f c) {
  // 8 args: (neg_a, A, neg_b, B, c_mod, C, reuse_a, reuse_b)
  return __builtin_amdgcn_wmma_f32_16x16x32_bf16(false, a, false, b, (short)0, c,
                                                 false, false);
}

// A fragment: 16(M) x 32(K) from row-major bf16, leading dim `ld` (elements).
// lane(m = lane&15, h = lane>>4): k = h*8 + (e&7) + (e&8)*2  -> two 16B chunks.
static __device__ __forceinline__ v16bf load_a_frag(const bf16_t* p, int ld) {
  int l = threadIdx.x & 31;
  int m = l & 15, h = l >> 4;
  const bf16_t* r = p + (size_t)m * ld + h * 8;
  v8bf lo = *(const v8bf*)(r);
  v8bf hi = *(const v8bf*)(r + 16);
  v16bf o;
#pragma unroll
  for (int i = 0; i < 8; ++i) { o[i] = lo[i]; o[i + 8] = hi[i]; }
  return o;
}

// B fragment: 32(K) x 16(N). Column n of B == row n of storage ([N][K] row-major,
// leading dim `ld`). lane(n = lane&15, h = lane>>4): k = h*8 + (e&7) + (e&8)*2  -> two 16B chunks.
static __device__ __forceinline__ v16bf load_b_frag(const bf16_t* p, int ld) {
  int l = threadIdx.x & 31;
  int n = l & 15, h = l >> 4;
  const bf16_t* r = p + (size_t)n * ld + h * 8;
  v8bf lo = *(const v8bf*)(r);
  v8bf hi = *(const v8bf*)(r + 16);
  v16bf o;
#pragma unroll
  for (int i = 0; i < 8; ++i) { o[i] = lo[i]; o[i + 8] = hi[i]; }
  return o;
}

// ---------------------------------------------------------------------------
// f32 -> bf16 conversion
// ---------------------------------------------------------------------------
__global__ __launch_bounds__(256) void cvt_bf16_k(const float* __restrict__ in,
                                                  bf16_t* __restrict__ out, int n) {
  int i = blockIdx.x * blockDim.x + threadIdx.x;
  int stride = gridDim.x * blockDim.x;
  for (; i < n; i += stride) out[i] = (bf16_t)in[i];
}

// ---------------------------------------------------------------------------
// GEMM: out = A(MxK, bf16 row-major) * Bt(NxK, bf16 row-major)^T
//   MODE 0: bf16 out, row-major [M][N]
//   MODE 1: bf16 out, transposed [N][M]   (used for V -> Vt)
//   MODE 2: f32 out, row-major [M][N], + bias[N]
// One wave computes a 64x64 tile: 4x4 WMMA accumulators (128 VGPRs),
// 16 WMMAs per 8 fragment loads in the K loop (L2-resident, compute-bound).
// ---------------------------------------------------------------------------
template <int MODE>
__global__ __launch_bounds__(256) void gemm_wmma_k(const bf16_t* __restrict__ A,
                                                   const bf16_t* __restrict__ Bt,
                                                   void* __restrict__ outp,
                                                   const float* __restrict__ bias,
                                                   int M, int N, int K) {
  int wid = blockIdx.x * (blockDim.x >> 5) + (threadIdx.x >> 5);
  int ntn = N >> 6;
  int total = (M >> 6) * ntn;
  if (wid >= total) return;                 // wave-uniform; EXEC stays all-1s
  int tm = wid / ntn, tn = wid % ntn;
  int l = threadIdx.x & 31;
  int ln = l & 15, lh = l >> 4;

  v8f c[4][4] = {};
  const bf16_t* Abase = A  + (size_t)(tm * 64) * K;
  const bf16_t* Bbase = Bt + (size_t)(tn * 64) * K;

  for (int kk = 0; kk < K; kk += 32) {
    v16bf a[4], b[4];
#pragma unroll
    for (int r = 0; r < 4; ++r) {
      a[r] = load_a_frag(Abase + (size_t)(r * 16) * K + kk, K);
      b[r] = load_b_frag(Bbase + (size_t)(r * 16) * K + kk, K);
    }
#pragma unroll
    for (int r = 0; r < 4; ++r)
#pragma unroll
      for (int cn = 0; cn < 4; ++cn)
        c[r][cn] = wmma_bf16(a[r], b[cn], c[r][cn]);
  }

#pragma unroll
  for (int im = 0; im < 4; ++im) {
#pragma unroll
    for (int jn = 0; jn < 4; ++jn) {
      int rowb = tm * 64 + im * 16 + lh * 8;  // rows rowb..rowb+7 (e index)
      int col  = tn * 64 + jn * 16 + ln;
      if (MODE == 0) {
        bf16_t* o = (bf16_t*)outp;
#pragma unroll
        for (int e = 0; e < 8; ++e)
          o[(size_t)(rowb + e) * N + col] = (bf16_t)c[im][jn][e];
      } else if (MODE == 1) {
        bf16_t* o = (bf16_t*)outp;           // [N][M]; 8 consecutive m -> 16B store
        v8bf v;
#pragma unroll
        for (int e = 0; e < 8; ++e) v[e] = (bf16_t)c[im][jn][e];
        *(v8bf*)(o + (size_t)col * M + rowb) = v;
      } else {
        float* o = (float*)outp;
        float bv = bias ? bias[col] : 0.0f;
#pragma unroll
        for (int e = 0; e < 8; ++e)
          o[(size_t)(rowb + e) * N + col] = c[im][jn][e] + bv;
      }
    }
  }
}

// ---------------------------------------------------------------------------
// Flash attention, causal. One wave per (head, 16-query block).
// Q,K: [S][D] bf16 row-major. Vt: [D][S] bf16 (per-head rows contiguous in S).
// Ctx out: [S][D] bf16 row-major.
// Wave -> qblk mapping pairs (i, 255-i) so adjacent waves' causal trip counts
// sum to a constant (load balance across WGPs).
// ---------------------------------------------------------------------------
__global__ __launch_bounds__(256) void attn_wmma_k(const bf16_t* __restrict__ Q,
                                                   const bf16_t* __restrict__ Kb,
                                                   const bf16_t* __restrict__ Vt,
                                                   bf16_t* __restrict__ Ctx) {
  __shared__ __align__(16) bf16_t Pls[8][16 * 40];  // per-wave 16x32 P tile, padded
  int wslot = threadIdx.x >> 5;
  int wid = blockIdx.x * 8 + wslot;
  int head = wid >> 8;            // S/16 = 256 q-blocks per head
  int qraw = wid & 255;
  int qblk = (qraw & 1) ? (255 - (qraw >> 1)) : (qraw >> 1);
  int qbase = qblk << 4;
  int hb = head * HD_DIM;
  int l = threadIdx.x & 31;
  int ln = l & 15, lh = l >> 4;

  // Q fragments for this 16-row block (d = 0..31 and 32..63), reused every iter.
  v16bf qa0 = load_a_frag(Q + (size_t)qbase * D_DIM + hb, D_DIM);
  v16bf qa1 = load_a_frag(Q + (size_t)qbase * D_DIM + hb + 32, D_DIM);

  v8f o0 = {}, o1 = {}, o2 = {}, o3 = {};   // 16 x 64 f32 context accumulator
  float mrow[8], lrow[8];
#pragma unroll
  for (int e = 0; e < 8; ++e) { mrow[e] = -3.0e38f; lrow[e] = 0.0f; }

  const float sc = 0.125f * 1.44269504088896340736f;  // 1/sqrt(64) * log2(e)

  for (int j = 0; j <= qbase + 15; j += 32) {
    // scores (log2 domain after scaling): tile a = keys j..j+15, b = j+16..j+31
    v8f sa = {}, sb = {};
    sa = wmma_bf16(qa0, load_b_frag(Kb + (size_t)j * D_DIM + hb, D_DIM), sa);
    sa = wmma_bf16(qa1, load_b_frag(Kb + (size_t)j * D_DIM + hb + 32, D_DIM), sa);
    sb = wmma_bf16(qa0, load_b_frag(Kb + (size_t)(j + 16) * D_DIM + hb, D_DIM), sb);
    sb = wmma_bf16(qa1, load_b_frag(Kb + (size_t)(j + 16) * D_DIM + hb + 32, D_DIM), sb);

    bool need_mask = (j + 31 > qbase);      // wave-uniform
    float pa[8], pb[8];
#pragma unroll
    for (int e = 0; e < 8; ++e) {
      float xa = sa[e] * sc;
      float xb = sb[e] * sc;
      if (need_mask) {
        int q = qbase + lh * 8 + e;
        xa = (j + ln      > q) ? -3.0e38f : xa;
        xb = (j + 16 + ln > q) ? -3.0e38f : xb;
      }
      pa[e] = xa; pb[e] = xb;
    }

    // online softmax per row (rows live in e; row spread across 16 lanes of half)
#pragma unroll
    for (int e = 0; e < 8; ++e) {
      float t = fmaxf(pa[e], pb[e]);
      t = fmaxf(t, __shfl_xor(t, 1, 16));
      t = fmaxf(t, __shfl_xor(t, 2, 16));
      t = fmaxf(t, __shfl_xor(t, 4, 16));
      t = fmaxf(t, __shfl_xor(t, 8, 16));
      float mnew = fmaxf(mrow[e], t);
      float corr = __builtin_amdgcn_exp2f(mrow[e] - mnew);
      float ea = __builtin_amdgcn_exp2f(pa[e] - mnew);
      float eb = __builtin_amdgcn_exp2f(pb[e] - mnew);
      float rs = ea + eb;
      rs += __shfl_xor(rs, 1, 16);
      rs += __shfl_xor(rs, 2, 16);
      rs += __shfl_xor(rs, 4, 16);
      rs += __shfl_xor(rs, 8, 16);
      lrow[e] = lrow[e] * corr + rs;
      mrow[e] = mnew;
      o0[e] *= corr; o1[e] *= corr; o2[e] *= corr; o3[e] *= corr;
      // C-layout -> LDS row-major P tile (row = lh*8+e, cols ln and 16+ln)
      Pls[wslot][(lh * 8 + e) * 40 + ln]      = (bf16_t)ea;
      Pls[wslot][(lh * 8 + e) * 40 + 16 + ln] = (bf16_t)eb;
    }
    asm volatile("s_wait_dscnt 0" ::: "memory");  // wave-private LDS RAW

    // reread P as A-fragment (16x32)
    v16bf pf;
    {
      const bf16_t* pr = &Pls[wslot][(l & 15) * 40 + lh * 8];
      v8bf lo = *(const v8bf*)(pr);
      v8bf hi = *(const v8bf*)(pr + 16);
#pragma unroll
      for (int i = 0; i < 8; ++i) { pf[i] = lo[i]; pf[i + 8] = hi[i]; }
    }
    asm volatile("s_wait_dscnt 0" ::: "memory");  // reads done before next iter's writes

    // ctx += P @ V  (Vt gives contiguous K per lane)
    o0 = wmma_bf16(pf, load_b_frag(Vt + (size_t)(hb +  0) * S_LEN + j, S_LEN), o0);
    o1 = wmma_bf16(pf, load_b_frag(Vt + (size_t)(hb + 16) * S_LEN + j, S_LEN), o1);
    o2 = wmma_bf16(pf, load_b_frag(Vt + (size_t)(hb + 32) * S_LEN + j, S_LEN), o2);
    o3 = wmma_bf16(pf, load_b_frag(Vt + (size_t)(hb + 48) * S_LEN + j, S_LEN), o3);
  }

  // normalize + store ctx bf16 row-major
#pragma unroll
  for (int e = 0; e < 8; ++e) {
    float inv = __builtin_amdgcn_rcpf(lrow[e]);
    size_t row = (size_t)(qbase + lh * 8 + e) * D_DIM + hb;
    Ctx[row + ln]      = (bf16_t)(o0[e] * inv);
    Ctx[row + 16 + ln] = (bf16_t)(o1[e] * inv);
    Ctx[row + 32 + ln] = (bf16_t)(o2[e] * inv);
    Ctx[row + 48 + ln] = (bf16_t)(o3[e] * inv);
  }
}

// ---------------------------------------------------------------------------
// Host launcher
// ---------------------------------------------------------------------------
extern "C" void kernel_launch(void* const* d_in, const int* in_sizes, int n_in,
                              void* d_out, int out_size, void* d_ws, size_t ws_size,
                              hipStream_t stream) {
  (void)in_sizes; (void)n_in; (void)out_size; (void)ws_size;
  const float* x  = (const float*)d_in[0];
  const float* Wq = (const float*)d_in[1];
  const float* Wk = (const float*)d_in[2];
  const float* Wv = (const float*)d_in[3];
  const float* Wo = (const float*)d_in[4];
  const float* bo = (const float*)d_in[5];
  float* out = (float*)d_out;

  char* ws = (char*)d_ws;
  size_t off = 0;
  auto take = [&](size_t elems) {
    bf16_t* p = (bf16_t*)(ws + off);
    off += ((elems * sizeof(bf16_t)) + 255) & ~(size_t)255;
    return p;
  };
  const size_t SD = (size_t)S_LEN * D_DIM;
  const size_t DD = (size_t)D_DIM * D_DIM;
  bf16_t* xb  = take(SD);
  bf16_t* wqb = take(DD);
  bf16_t* wkb = take(DD);
  bf16_t* wvb = take(DD);
  bf16_t* wob = take(DD);
  bf16_t* Qb  = take(SD);
  bf16_t* Kb  = take(SD);
  bf16_t* Vt  = take(SD);   // transposed [D][S]
  bf16_t* Ctx = take(SD);

  cvt_bf16_k<<<512, 256, 0, stream>>>(x,  xb,  (int)SD);
  cvt_bf16_k<<<256, 256, 0, stream>>>(Wq, wqb, (int)DD);
  cvt_bf16_k<<<256, 256, 0, stream>>>(Wk, wkb, (int)DD);
  cvt_bf16_k<<<256, 256, 0, stream>>>(Wv, wvb, (int)DD);
  cvt_bf16_k<<<256, 256, 0, stream>>>(Wo, wob, (int)DD);

  // (S/64)*(D/64) = 1024 waves -> 128 blocks of 8 waves
  int gemm_blocks = ((S_LEN / 64) * (D_DIM / 64) + 7) / 8;
  gemm_wmma_k<0><<<gemm_blocks, 256, 0, stream>>>(xb, wqb, Qb, nullptr, S_LEN, D_DIM, D_DIM);
  gemm_wmma_k<0><<<gemm_blocks, 256, 0, stream>>>(xb, wkb, Kb, nullptr, S_LEN, D_DIM, D_DIM);
  gemm_wmma_k<1><<<gemm_blocks, 256, 0, stream>>>(xb, wvb, Vt, nullptr, S_LEN, D_DIM, D_DIM);

  // H * S/16 = 4096 waves -> 512 blocks
  attn_wmma_k<<<512, 256, 0, stream>>>(Qb, Kb, Vt, Ctx);

  gemm_wmma_k<2><<<gemm_blocks, 256, 0, stream>>>(Ctx, wob, out, bo, S_LEN, D_DIM, D_DIM);
}
